// StagModelContrastive_54623394070984
// MI455X (gfx1250) — matrix operand order repaired and checked by reading, not compiled, not run on hardware
//
#include <hip/hip_runtime.h>
#include <hip/hip_bf16.h>

#define NN 50000
#define NE 800000
#define DD 64
#define ETILES (NE / 16)
#define NTILES (NN / 16)

typedef __attribute__((ext_vector_type(16))) _Float16 v16h;
typedef __attribute__((ext_vector_type(8)))  _Float16 v8h;
typedef __attribute__((ext_vector_type(8)))  float    v8f;

union AFrag { v16h v; v8h h[2]; };
union H2 { _Float16 h[2]; unsigned int u; };

__device__ __forceinline__ v8f wmma16(v16h a, v16h b, v8f c) {
  return __builtin_amdgcn_wmma_f32_16x16x32_f16(false, a, false, b,
                                                (short)0, c, false, false);
}

// Compiler scheduling fence only: DS ops from the same wave complete in order
// (ISA 7.3), so a ds_load after ds_stores to the same wave-private buffer
// already sees the data; no hardware dscnt drain needed here.
__device__ __forceinline__ void lds_handoff() {
  __builtin_amdgcn_wave_barrier();
}

// relu without the fmaxf canonicalization double-max (WMMA output is not sNaN)
__device__ __forceinline__ float relu(float x) {
  float r;
  asm("v_max_num_f32 %0, 0, %1" : "=v"(r) : "v"(x));
  return r;
}

__device__ __forceinline__ unsigned int pack2(float x, float y) {
  H2 p;
  p.h[0] = (_Float16)x;
  p.h[1] = (_Float16)y;
  return p.u;
}

__device__ __forceinline__ v8f bcast8(float v) {
  v8f t;
#pragma unroll
  for (int r = 0; r < 8; r++) t[r] = v;
  return t;
}

// ---------------------------------------------------------------------------
// Edge MLP pass: he = relu([h[src],h[dst]] @ We + be); loc = he.Wloc + bloc;
// optionally ls = he.Wls + bls.  SCATTER: agg[dst] += h[src]*loc.
// NLL: accumulate -logN(xt; loc, exp(ls)) summed over edges into *nll.
// ---------------------------------------------------------------------------
template <bool SCATTER, bool NLL>
__global__ __launch_bounds__(256) void edge_pass(
    const float* __restrict__ h, const int* __restrict__ src,
    const int* __restrict__ dst, const float* __restrict__ We,
    const float* __restrict__ be, const float* __restrict__ Wloc,
    const float* __restrict__ bloc, const float* __restrict__ Wls,
    const float* __restrict__ bls, float* __restrict__ agg,
    float* __restrict__ nll, float xt, int ntiles) {
  __shared__ _Float16 sWe[8192];      // 16 fragments * 32 lanes * 16 halves
  __shared__ _Float16 sX[8][2048];    // per-wave 16 x 128 halves

  const int tid = threadIdx.x;
  const int lane = tid & 31;
  const int wv = tid >> 5;

  // Stage We (128x64, row-major) swizzled into B-fragment order:
  // fragment f=(c*4+n): lane L holds halves i -> K=32c+i+16*(L>>4), N=16n+(L&15)
  for (int idx = tid; idx < 8192; idx += 256) {
    int i = idx & 15;
    int L = (idx >> 4) & 31;
    int f = idx >> 9;
    int c = f >> 2, n = f & 3;
    int K = 32 * c + i + 16 * (L >> 4);
    int N = 16 * n + (L & 15);
    sWe[idx] = (_Float16)We[K * 64 + N];
  }
  __syncthreads();

  // Hoist all 16 B fragments into registers (loop-invariant).
  v16h Bf[16];
#pragma unroll
  for (int f = 0; f < 16; f++) {
    Bf[f] = *(const v16h*)(sWe + (f * 32 + lane) * 16);
  }

  const int m16 = lane & 15;
  const int g = lane >> 4;

  float ben[4], wl[4], wsc[4];
#pragma unroll
  for (int n = 0; n < 4; n++) {
    ben[n] = be[16 * n + m16];
    wl[n] = Wloc[16 * n + m16];
    wsc[n] = NLL ? Wls[16 * n + m16] : 0.0f;
  }
  const float blocv = bloc[0];
  const float blsv = NLL ? bls[0] : 0.0f;

  float nllacc = 0.0f;
  _Float16* X = sX[wv];
  const int nwaves = gridDim.x * 8;

  for (int tile = blockIdx.x * 8 + wv; tile < ntiles; tile += nwaves) {
    const int e0 = tile * 16;
    const int sm = src[e0 + m16];
    const int dm = dst[e0 + m16];

    // Gather: half-wave per edge; group g owns edge mm+8g, 4 cols per lane.
    // One b128 global load + one b64 ds store per 64-wide row-half.
#pragma unroll
    for (int mm = 0; mm < 8; mm++) {
      const int m = mm + 8 * g;
      int s = __shfl(sm, m, 32);
      int d = __shfl(dm, m, 32);
      float4 a = ((const float4*)(h + s * 64))[m16];
      float4 b = ((const float4*)(h + d * 64))[m16];
      uint2 ua = make_uint2(pack2(a.x, a.y), pack2(a.z, a.w));
      uint2 ub = make_uint2(pack2(b.x, b.y), pack2(b.z, b.w));
      ((uint2*)(X + m * 128))[m16] = ua;
      ((uint2*)(X + m * 128 + 64))[m16] = ub;
    }
    lds_handoff();

    // Bias folded into accumulator init: C[n] starts at be broadcast.
    v8f C[4];
#pragma unroll
    for (int n = 0; n < 4; n++) C[n] = bcast8(ben[n]);

    const _Float16* Arow = X + m16 * 128;
#pragma unroll
    for (int c = 0; c < 4; c++) {
      AFrag a;
      a.h[0] = *(const v8h*)(Arow + 32 * c + 16 * g);
      a.h[1] = *(const v8h*)(Arow + 32 * c + 16 * g + 8);
#pragma unroll
      for (int n = 0; n < 4; n++) {
        C[n] = wmma16(a.v, Bf[c * 4 + n], C[n]);
      }
    }

    // Heads: per-row partial dots, reduced across the 16-lane half-group.
    // After the xor tree, every lane in group g holds row (r + 8g) sums.
    float plr[8], psr[8];
#pragma unroll
    for (int r = 0; r < 8; r++) { plr[r] = 0.0f; psr[r] = 0.0f; }
#pragma unroll
    for (int n = 0; n < 4; n++) {
#pragma unroll
      for (int r = 0; r < 8; r++) {
        float hh = relu(C[n][r]);
        plr[r] += hh * wl[n];
        if (NLL) psr[r] += hh * wsc[n];
      }
    }
#pragma unroll
    for (int r = 0; r < 8; r++) {
#pragma unroll
      for (int off = 8; off > 0; off >>= 1) {
        plr[r] += __shfl_xor(plr[r], off, 32);
        if (NLL) psr[r] += __shfl_xor(psr[r], off, 32);
      }
      plr[r] += blocv;
      if (NLL) psr[r] += blsv;
    }

    if (NLL) {
      if (m16 == 0) {  // lanes 0 (rows 0..7) and 16 (rows 8..15)
#pragma unroll
        for (int r = 0; r < 8; r++) {
          float z = (xt - plr[r]) * __expf(-psr[r]);
          nllacc += 0.5f * z * z + psr[r] + 0.91893853320467274f;
        }
      }
    }

    if (SCATTER) {
      // Half-wave per edge: loc is already resident in plr[mm] for the
      // owning group -- no broadcast shuffles needed.
#pragma unroll
      for (int mm = 0; mm < 8; mm++) {
        const int m = mm + 8 * g;
        float locm = plr[mm];
        int s = __shfl(sm, m, 32);
        int d = __shfl(dm, m, 32);
        float4 f = ((const float4*)(h + s * 64))[m16];
        float* ap = agg + d * 64 + 4 * m16;
        atomicAdd(ap + 0, f.x * locm);
        atomicAdd(ap + 1, f.y * locm);
        atomicAdd(ap + 2, f.z * locm);
        atomicAdd(ap + 3, f.w * locm);
      }
    }
  }

  if (NLL) {
    nllacc += __shfl_xor(nllacc, 16, 32);
    if (lane == 0) atomicAdd(nll, nllacc);
  }
}

// ---------------------------------------------------------------------------
// Node update: out = relu((acc * norm) @ W + b)   (NNx64 @ 64x64 GEMM)
// ---------------------------------------------------------------------------
__global__ __launch_bounds__(256) void node_update(
    const float* __restrict__ acc, const float* __restrict__ nrm,
    const float* __restrict__ W, const float* __restrict__ b,
    float* __restrict__ out, int ntiles) {
  __shared__ _Float16 sW[4096];     // 8 fragments * 32 * 16
  __shared__ _Float16 sX[8][1024];  // per-wave 16 x 64 halves

  const int tid = threadIdx.x;
  const int lane = tid & 31;
  const int wv = tid >> 5;

  for (int idx = tid; idx < 4096; idx += 256) {
    int i = idx & 15;
    int L = (idx >> 4) & 31;
    int f = idx >> 9;
    int c = f >> 2, n = f & 3;
    int K = 32 * c + i + 16 * (L >> 4);
    int N = 16 * n + (L & 15);
    sW[idx] = (_Float16)W[K * 64 + N];
  }
  __syncthreads();

  v16h Bf[8];
#pragma unroll
  for (int f = 0; f < 8; f++) {
    Bf[f] = *(const v16h*)(sW + (f * 32 + lane) * 16);
  }

  const int m16 = lane & 15;
  const int g = lane >> 4;
  float bn[4];
#pragma unroll
  for (int n = 0; n < 4; n++) bn[n] = b[16 * n + m16];

  _Float16* X = sX[wv];
  const int nwaves = gridDim.x * 8;

  for (int tile = blockIdx.x * 8 + wv; tile < ntiles; tile += nwaves) {
    const int r0 = tile * 16;
    // Half-wave per row; 4 cols per lane (b128 load + b64 ds store).
#pragma unroll
    for (int mm = 0; mm < 8; mm++) {
      const int m = mm + 8 * g;
      const int row = r0 + m;
      float nv = nrm[row];
      float4 a = ((const float4*)(acc + row * 64))[m16];
      uint2 u = make_uint2(pack2(a.x * nv, a.y * nv), pack2(a.z * nv, a.w * nv));
      ((uint2*)(X + m * 64))[m16] = u;
    }
    lds_handoff();

    v8f C[4];
#pragma unroll
    for (int n = 0; n < 4; n++) C[n] = bcast8(bn[n]);

    const _Float16* Arow = X + m16 * 64;
#pragma unroll
    for (int c = 0; c < 2; c++) {
      AFrag a;
      a.h[0] = *(const v8h*)(Arow + 32 * c + 16 * g);
      a.h[1] = *(const v8h*)(Arow + 32 * c + 16 * g + 8);
#pragma unroll
      for (int n = 0; n < 4; n++) {
        C[n] = wmma16(a.v, Bf[c * 4 + n], C[n]);
      }
    }
#pragma unroll
    for (int n = 0; n < 4; n++) {
#pragma unroll
      for (int r = 0; r < 8; r++) {
        out[(r0 + r + 8 * g) * 64 + 16 * n + m16] = relu(C[n][r]);
      }
    }
  }
}

__global__ void deg_kernel(const int* __restrict__ dst, float* __restrict__ deg,
                           int E) {
  int e = blockIdx.x * 256 + threadIdx.x;
  if (e < E) atomicAdd(&deg[dst[e]], 1.0f);
}

__global__ void invert_kernel(float* __restrict__ deg, int n) {
  int i = blockIdx.x * 256 + threadIdx.x;
  if (i < n) deg[i] = 1.0f / fmaxf(deg[i], 1.0f);
}

__global__ void finalize_kernel(const float* __restrict__ nll,
                                float* __restrict__ out) {
  out[NN * DD] = nll[0] * (1.0f / (float)NE);
}

extern "C" void kernel_launch(void* const* d_in, const int* in_sizes, int n_in,
                              void* d_out, int out_size, void* d_ws,
                              size_t ws_size, hipStream_t stream) {
  const float* feat = (const float*)d_in[0];
  const int* src = (const int*)d_in[1];
  const int* dst = (const int*)d_in[2];
  const int* fsrc = (const int*)d_in[3];
  const int* fdst = (const int*)d_in[4];
  const float* W1 = (const float*)d_in[5];
  const float* b1 = (const float*)d_in[6];
  const float* W2 = (const float*)d_in[7];
  const float* b2 = (const float*)d_in[8];
  const float* We = (const float*)d_in[9];
  const float* be = (const float*)d_in[10];
  const float* Wloc = (const float*)d_in[11];
  const float* bloc = (const float*)d_in[12];
  const float* Wls = (const float*)d_in[13];
  const float* bls = (const float*)d_in[14];

  float* out = (float*)d_out;
  float* ws = (float*)d_ws;
  float* h1 = ws;                        // NN*DD
  float* agg = ws + (size_t)NN * DD;     // NN*DD
  float* nrm = ws + (size_t)2 * NN * DD; // NN
  float* nll = nrm + NN;                 // 1 (+pad)

  hipMemsetAsync(agg, 0, (size_t)NN * DD * sizeof(float), stream);
  hipMemsetAsync(nrm, 0, (size_t)NN * sizeof(float), stream);
  hipMemsetAsync(nll, 0, 16, stream);

  // degree + norm
  deg_kernel<<<(NE + 255) / 256, 256, 0, stream>>>(dst, nrm, NE);
  invert_kernel<<<(NN + 255) / 256, 256, 0, stream>>>(nrm, NN);

  // Layer 1: edge weights + aggregate (layer-1 NLL is overwritten in ref)
  edge_pass<true, false><<<1250, 256, 0, stream>>>(
      feat, src, dst, We, be, Wloc, bloc, Wls, bls, agg, nullptr, 1.0f, ETILES);
  node_update<<<391, 256, 0, stream>>>(agg, nrm, W1, b1, h1, NTILES);

  // Layer 2: pos edges (scatter + NLL vs x=1), fake edges (NLL vs x=0)
  hipMemsetAsync(agg, 0, (size_t)NN * DD * sizeof(float), stream);
  edge_pass<true, true><<<1250, 256, 0, stream>>>(
      h1, src, dst, We, be, Wloc, bloc, Wls, bls, agg, nll, 1.0f, ETILES);
  edge_pass<false, true><<<1250, 256, 0, stream>>>(
      h1, fsrc, fdst, We, be, Wloc, bloc, Wls, bls, nullptr, nll, 0.0f, ETILES);
  node_update<<<391, 256, 0, stream>>>(agg, nrm, W2, b2, out, NTILES);

  finalize_kernel<<<1, 1, 0, stream>>>(nll, out);
}
